// DGCNN_seg_78194174591134
// MI455X (gfx1250) — matrix operand order repaired and checked by reading, not compiled, not run on hardware
//
#include <hip/hip_runtime.h>
#include <cstdint>
#include <cstddef>

// ---------------------------------------------------------------------------
// DGCNN part-seg forward for MI455X (gfx1250, wave32, WMMA).
// All GEMMs (1x1 convs + per-batch Gram matrices for kNN) run through a
// wave32 v_wmma_f32_16x16x32_f16 kernel (16x64 output per wave: one A
// fragment reused across 4 B fragments) with f32 accumulation and fused
// epilogues (BN scale/bias + leaky-relu, Gram bias, final bias->f32).
// Activations are stored f16 (memory-bound workload @ 23.3 TB/s HBM);
// the 64MB per-batch Gram matrix stays L2-resident (192 MB L2).
// ---------------------------------------------------------------------------

typedef _Float16 f16;
typedef _Float16 h8   __attribute__((ext_vector_type(8)));
typedef _Float16 v16h __attribute__((ext_vector_type(16)));
typedef float    v8f  __attribute__((ext_vector_type(8)));

#define KNN 20
#define NP  4096
#define NB  4
#define PTS (NB * NP)       // 16384 points total
#define XNK (PTS * KNN)     // 327680 point-neighbor rows

__device__ __forceinline__ float lrelu(float v) { return v > 0.f ? v : 0.2f * v; }

// ---------------------------------------------------------------------------
// Generic WMMA GEMM: out[M, Ncols] = A[M, Kdim] * W[Ncols, Kdim]^T
// A: f16 row-major, lda (mult of 8). W: packed f16 row-major, ldw (mult of 8).
// grid = (M/16, Ncols/(16*NT)), block = 32 (one wave -> 16 x 16*NT tile).
// The A fragment is loaded once per K-step and reused across NT WMMAs.
// mode 0: out f16 = lrelu(acc*s[col] + b[col])
// mode 1: out f32 = acc*(-2) + p0[col]                     (Gram / distance)
// mode 2: out f32 d_out[(b*2+col)*npts + n] = acc + b[col], col<2 only
// ---------------------------------------------------------------------------
template <int NT>
__global__ __launch_bounds__(32)
void wmma_gemm_kernel(const f16* __restrict__ A, int lda,
                      const f16* __restrict__ W, int ldw,
                      int Kdim,
                      const float* __restrict__ p0,
                      const float* __restrict__ p1,
                      int mode,
                      f16* __restrict__ outH,
                      float* __restrict__ outF,
                      int ldo, int npts)
{
    const int lane = threadIdx.x;              // 0..31
    const int tM = blockIdx.x;
    const int colBase = blockIdx.y * (16 * NT);

    // A fragment: lane holds row M = lane&15; halves 0..7 -> K = ksh+0..7,
    // halves 8..15 -> K = 16+ksh+0..7, ksh = 8*(lane>=16).  (ISA 7.12.2)
    const int mrow = tM * 16 + (lane & 15);
    const int ksh  = (lane >> 4) * 8;
    // B fragment j: lane holds col N = lane&15 of W row (colBase+16j+lane&15);
    // halves 0..7 -> K = kbsh+0..7, halves 8..15 -> K = 16+kbsh+0..7, kbsh = 8*(lane>=16).
    const int kbsh = (lane >> 4) * 8;

    const f16* Ap = A + (size_t)mrow * lda + ksh;
    const f16* Wp = W + (size_t)(colBase + (lane & 15)) * ldw + kbsh;

    v8f acc[NT];
#pragma unroll
    for (int j = 0; j < NT; ++j) acc[j] = (v8f){};

    for (int kk = 0; kk < Kdim; kk += 32) {
        h8 a0 = *(const h8*)(Ap + kk);
        h8 a1 = *(const h8*)(Ap + kk + 16);
        v16h av = __builtin_shufflevector(a0, a1, 0,1,2,3,4,5,6,7,8,9,10,11,12,13,14,15);
#pragma unroll
        for (int j = 0; j < NT; ++j) {
            const f16* wj = Wp + (size_t)j * 16 * ldw + kk;
            h8 b0 = *(const h8*)(wj);
            h8 b1 = *(const h8*)(wj + 16);
            v16h bv = __builtin_shufflevector(b0, b1, 0,1,2,3,4,5,6,7,8,9,10,11,12,13,14,15);
            acc[j] = __builtin_amdgcn_wmma_f32_16x16x32_f16(
                         false, av, false, bv, (short)0, acc[j], false, false);
        }
    }

    // C/D layout: VGPR r -> M = r + 8*(lane>=16), N = lane&15  (ISA 7.12.2)
    const int mbase = tM * 16 + ((lane >> 4) * 8);

#pragma unroll
    for (int j = 0; j < NT; ++j) {
        const int col = colBase + j * 16 + (lane & 15);
        if (mode == 0) {
            const float s = p0[col], b = p1[col];
#pragma unroll
            for (int r = 0; r < 8; ++r) {
                float v = lrelu(acc[j][r] * s + b);
                outH[(size_t)(mbase + r) * ldo + col] = (f16)v;
            }
        } else if (mode == 1) {
            const float cb = p0[col];
#pragma unroll
            for (int r = 0; r < 8; ++r)
                outF[(size_t)(mbase + r) * ldo + col] = acc[j][r] * -2.f + cb;
        } else {
            if (col < 2) {
                const float b = p1[col];
#pragma unroll
                for (int r = 0; r < 8; ++r) {
                    int m = mbase + r;
                    int bi = m / npts, n = m - bi * npts;
                    outF[((size_t)bi * 2 + col) * npts + n] = acc[j][r] + b;
                }
            }
        }
    }
}

// ---------------------------------------------------------------------------
// Weight packing: f32 [Co,Ci] -> f16 [CoP,CiP], zero padded.
// ---------------------------------------------------------------------------
__global__ void pack_kernel(const float* __restrict__ W, int Co, int Ci,
                            f16* __restrict__ Wp, int CoP, int CiP, int total)
{
    int t = blockIdx.x * blockDim.x + threadIdx.x;
    if (t >= total) return;
    int r = t / CiP, c = t - r * CiP;
    Wp[t] = (r < Co && c < Ci) ? (f16)W[r * Ci + c] : (f16)0.f;
}

// ---------------------------------------------------------------------------
// kNN on 3-D points (one thread per query, LDS candidate tiles).
// Keeps 20 smallest squared distances, ascending, stable ties (matches
// lax.top_k on pd = -dist^2, which includes self at rank 0).
// ---------------------------------------------------------------------------
__global__ __launch_bounds__(128)
void knn3_kernel(const float* __restrict__ pts, int* __restrict__ idx)
{
    __shared__ float sx[128], sy[128], sz[128];
    int gid = blockIdx.x * 128 + threadIdx.x;     // global point over B*N
    int b = gid / NP;
    int n = gid - b * NP;
    const float* base = pts + (size_t)b * NP * 3;
    float qx = base[n * 3 + 0], qy = base[n * 3 + 1], qz = base[n * 3 + 2];

    float bd[KNN]; int bi[KNN];
#pragma unroll
    for (int t = 0; t < KNN; ++t) { bd[t] = 3.4e38f; bi[t] = 0; }

    for (int j0 = 0; j0 < NP; j0 += 128) {
        __syncthreads();
        int j = j0 + threadIdx.x;
        sx[threadIdx.x] = base[j * 3 + 0];
        sy[threadIdx.x] = base[j * 3 + 1];
        sz[threadIdx.x] = base[j * 3 + 2];
        __syncthreads();
        for (int jj = 0; jj < 128; ++jj) {
            float dx = sx[jj] - qx, dy = sy[jj] - qy, dz = sz[jj] - qz;
            float d = dx * dx + dy * dy + dz * dz;
            int cand = j0 + jj;
            if (d < bd[KNN - 1]) {
#pragma unroll
                for (int t = KNN - 1; t >= 0; --t) {
                    float pdv = (t > 0) ? bd[t - 1] : -1.f;
                    int   piv = (t > 0) ? bi[t - 1] : 0;
                    if (d < bd[t]) {
                        if (t > 0 && d < pdv) { bd[t] = pdv; bi[t] = piv; }
                        else                  { bd[t] = d;   bi[t] = cand; }
                    }
                }
            }
        }
    }
#pragma unroll
    for (int t = 0; t < KNN; ++t) idx[(size_t)gid * KNN + t] = bi[t];
}

// Top-20 smallest per row of the f32 Gram/distance matrix (one batch).
__global__ __launch_bounds__(128)
void topk_gram_kernel(const float* __restrict__ gram, int* __restrict__ idx)
{
    int n = blockIdx.x * 128 + threadIdx.x;
    const float* row = gram + (size_t)n * NP;
    float bd[KNN]; int bi[KNN];
#pragma unroll
    for (int t = 0; t < KNN; ++t) { bd[t] = 3.4e38f; bi[t] = 0; }
    for (int j = 0; j < NP; ++j) {
        float d = row[j];
        if (d < bd[KNN - 1]) {
#pragma unroll
            for (int t = KNN - 1; t >= 0; --t) {
                float pdv = (t > 0) ? bd[t - 1] : -3.4e38f;
                int   piv = (t > 0) ? bi[t - 1] : 0;
                if (d < bd[t]) {
                    if (t > 0 && d < pdv) { bd[t] = pdv; bi[t] = piv; }
                    else                  { bd[t] = d;   bi[t] = j; }
                }
            }
        }
    }
#pragma unroll
    for (int t = 0; t < KNN; ++t) idx[(size_t)n * KNN + t] = bi[t];
}

// Edge feature from 3-D coords -> [X, 32] f16 (cols 0..2 diff, 3..5 ctr, pad).
__global__ void edge6_kernel(const float* __restrict__ pts,
                             const int* __restrict__ idx, f16* __restrict__ e)
{
    int x = blockIdx.x * blockDim.x + threadIdx.x;      // over XNK
    int k = x % KNN, p = x / KNN;
    int b = p / NP,  n = p - b * NP;
    const float* base = pts + (size_t)b * NP * 3;
    int j = idx[(size_t)p * KNN + k];
    f16* er = e + (size_t)x * 32;
    float cx = base[n * 3], cy = base[n * 3 + 1], cz = base[n * 3 + 2];
    er[0] = (f16)(base[j * 3 + 0] - cx);
    er[1] = (f16)(base[j * 3 + 1] - cy);
    er[2] = (f16)(base[j * 3 + 2] - cz);
    er[3] = (f16)cx; er[4] = (f16)cy; er[5] = (f16)cz;
#pragma unroll
    for (int c = 6; c < 32; ++c) er[c] = (f16)0.f;
}

// Edge feature from f16 features (C channels) -> [X, 2C] f16.
__global__ void edgeF_kernel(const f16* __restrict__ feat, int ldf, int C,
                             const int* __restrict__ idx, f16* __restrict__ e, int lde)
{
    int t = blockIdx.x * blockDim.x + threadIdx.x;      // over XNK * C
    int c = t % C, x = t / C;
    int k = x % KNN, p = x / KNN;
    int b = p / NP;
    int j = idx[(size_t)p * KNN + k] + b * NP;
    float fc = (float)feat[(size_t)p * ldf + c];
    float fn = (float)feat[(size_t)j * ldf + c];
    e[(size_t)x * lde + c]     = (f16)(fn - fc);
    e[(size_t)x * lde + C + c] = (f16)fc;
}

// Max over the 20 neighbors: [P*KNN, C] -> [P, C].
__global__ void maxk_kernel(const f16* __restrict__ in, int ldi,
                            f16* __restrict__ out, int ldo, int C)
{
    int t = blockIdx.x * blockDim.x + threadIdx.x;      // over P*C
    int c = t % C, p = t / C;
    float m = -3.4e38f;
#pragma unroll 4
    for (int k = 0; k < KNN; ++k)
        m = fmaxf(m, (float)in[((size_t)p * KNN + k) * ldi + c]);
    out[(size_t)p * ldo + c] = (f16)m;
}

// Max over N per batch: f16 [B*N, ldi] -> f32 [B, C].
__global__ void colmax_kernel(const f16* __restrict__ in, int ldi,
                              float* __restrict__ out, int C)
{
    int t = blockIdx.x * blockDim.x + threadIdx.x;      // over NB*C
    int b = t / C, c = t - b * C;
    const f16* base = in + (size_t)b * NP * ldi + c;
    float m = -3.4e38f;
    for (int n = 0; n < NP; ++n) m = fmaxf(m, (float)base[(size_t)n * ldi]);
    out[t] = m;
}

// Squared norms per point (for Gram -> distance epilogue).
__global__ void sqnorm_kernel(const f16* __restrict__ feat, int ldf, int C,
                              float* __restrict__ sq)
{
    int p = blockIdx.x * blockDim.x + threadIdx.x;      // over PTS
    float s = 0.f;
    for (int c = 0; c < C; ++c) { float v = (float)feat[(size_t)p * ldf + c]; s += v * v; }
    sq[p] = s;
}

// Tiny FC: out[b,o] = [lrelu]((in[b,:].W[o,:])*s[o] + bias[o]).
__global__ void fc_kernel(const float* __restrict__ in, int K,
                          const float* __restrict__ W,
                          const float* __restrict__ s,
                          const float* __restrict__ bias,
                          float* __restrict__ out, int O, int relu, int total)
{
    int t = blockIdx.x * blockDim.x + threadIdx.x;
    if (t >= total) return;
    int b = t / O, o = t - b * O;
    float a = 0.f;
    for (int k = 0; k < K; ++k) a += in[b * K + k] * W[o * K + k];
    if (s) a *= s[o];
    a += bias[o];
    if (relu) a = lrelu(a);
    out[t] = a;
}

// Apply learned 3x3 transform: xt[b,n,d] = sum_c x[b,n,c]*T[b,c,d].
__global__ void xform_kernel(const float* __restrict__ x,
                             const float* __restrict__ t9,
                             float* __restrict__ xt)
{
    int g = blockIdx.x * blockDim.x + threadIdx.x;      // over PTS
    int b = g / NP;
    const float* xr = x + (size_t)g * 3;
    const float* T  = t9 + b * 9;
#pragma unroll
    for (int d = 0; d < 3; ++d)
        xt[(size_t)g * 3 + d] = xr[0] * T[d] + xr[1] * T[3 + d] + xr[2] * T[6 + d];
}

// Categorical-vector branch: lrelu((cvec*catw + catlb)*cats + catbb) -> [B,64].
__global__ void cv_kernel(const float* __restrict__ cvec,
                          const float* __restrict__ catw,
                          const float* __restrict__ catlb,
                          const float* __restrict__ cats,
                          const float* __restrict__ catbb,
                          float* __restrict__ cv)
{
    int t = threadIdx.x;                                 // 256 = NB*64
    int b = t / 64, o = t - b * 64;
    float v = (cvec[b] * catw[o] + catlb[o]) * cats[o] + catbb[o];
    cv[t] = lrelu(v);
}

// cat = [x1 | x2 | x3] -> [P, 256] f16.
__global__ void cat_kernel(const f16* __restrict__ x1, const f16* __restrict__ x2,
                           const f16* __restrict__ x3, f16* __restrict__ cat)
{
    int t = blockIdx.x * blockDim.x + threadIdx.x;      // over PTS*256
    int c = t % 256, p = t / 256;
    f16 v = (c < 64)  ? x1[(size_t)p * 64 + c]
          : (c < 128) ? x2[(size_t)p * 64 + (c - 64)]
                      : x3[(size_t)p * 128 + (c - 128)];
    cat[t] = v;
}

// h = [xg(1024) | cv(64) | x4(1024)] -> [P, 2112] f16.
__global__ void hcat_kernel(const float* __restrict__ xg, const float* __restrict__ cv,
                            const f16* __restrict__ x4, f16* __restrict__ h)
{
    int t = blockIdx.x * blockDim.x + threadIdx.x;      // over PTS*2112
    int c = t % 2112, p = t / 2112;
    int b = p / NP;
    f16 v;
    if (c < 1024)      v = (f16)xg[b * 1024 + c];
    else if (c < 1088) v = (f16)cv[b * 64 + (c - 1024)];
    else               v = x4[(size_t)p * 1024 + (c - 1088)];
    h[(size_t)p * 2112 + c] = v;
}

// ---------------------------------------------------------------------------
// Host-side launch helpers
// ---------------------------------------------------------------------------
static inline void gemm(hipStream_t s, const f16* A, int lda, const f16* W, int ldw,
                        int M, int N, int K, const float* p0, const float* p1,
                        int mode, f16* oH, float* oF, int ldo, int npts)
{
    if (N % 64 == 0) {
        dim3 g(M / 16, N / 64);
        wmma_gemm_kernel<4><<<g, 32, 0, s>>>(A, lda, W, ldw, K, p0, p1, mode, oH, oF, ldo, npts);
    } else {
        dim3 g(M / 16, N / 16);
        wmma_gemm_kernel<1><<<g, 32, 0, s>>>(A, lda, W, ldw, K, p0, p1, mode, oH, oF, ldo, npts);
    }
}

static inline void packW(hipStream_t s, const float* W, int Co, int Ci,
                         f16* Wp, int CoP, int CiP)
{
    int n = CoP * CiP;
    pack_kernel<<<(n + 255) / 256, 256, 0, s>>>(W, Co, Ci, Wp, CoP, CiP, n);
}

extern "C" void kernel_launch(void* const* d_in, const int* in_sizes, int n_in,
                              void* d_out, int out_size, void* d_ws, size_t ws_size,
                              hipStream_t stream)
{
    (void)in_sizes; (void)n_in; (void)out_size; (void)ws_size;
    const float* x    = (const float*)d_in[0];
    const float* cvec = (const float*)d_in[1];
    const float* tw1 = (const float*)d_in[2],  *ts1 = (const float*)d_in[3],  *tb1 = (const float*)d_in[4];
    const float* tw2 = (const float*)d_in[5],  *ts2 = (const float*)d_in[6],  *tb2 = (const float*)d_in[7];
    const float* tw3 = (const float*)d_in[8],  *ts3 = (const float*)d_in[9],  *tb3 = (const float*)d_in[10];
    const float* tl1 = (const float*)d_in[11], *tls1 = (const float*)d_in[12], *tlb1 = (const float*)d_in[13];
    const float* tl2 = (const float*)d_in[14], *tls2 = (const float*)d_in[15], *tlb2 = (const float*)d_in[16];
    const float* ttw = (const float*)d_in[17], *ttb = (const float*)d_in[18];
    const float* w1  = (const float*)d_in[19], *s1 = (const float*)d_in[20], *b1 = (const float*)d_in[21];
    const float* w2  = (const float*)d_in[22], *s2 = (const float*)d_in[23], *b2 = (const float*)d_in[24];
    const float* w3  = (const float*)d_in[25], *s3 = (const float*)d_in[26], *b3 = (const float*)d_in[27];
    const float* w5  = (const float*)d_in[28], *s5 = (const float*)d_in[29], *b5 = (const float*)d_in[30];
    const float* w6  = (const float*)d_in[31], *s6 = (const float*)d_in[32], *b6 = (const float*)d_in[33];
    const float* w7  = (const float*)d_in[34], *s7 = (const float*)d_in[35], *b7 = (const float*)d_in[36];
    const float* w8  = (const float*)d_in[37], *b8 = (const float*)d_in[38];
    const float* catw = (const float*)d_in[39], *catlb = (const float*)d_in[40];
    const float* cats = (const float*)d_in[41], *catbb = (const float*)d_in[42];
    float* out = (float*)d_out;

    // ---- workspace layout (~218 MB, 256B aligned blocks) ----
    char* ws = (char*)d_ws;
    size_t o = 0;
    f16* WP   = (f16*)(ws + o); o += (size_t)4 << 20;           // packed weights
    f16* EBUF = (f16*)(ws + o); o += (size_t)XNK * 128 * 2;     // edge feats / hcat
    f16* HBUF = (f16*)(ws + o); o += (size_t)XNK * 128 * 2;     // conv out / gram / h6/h7
    f16* HM   = (f16*)(ws + o); o += (size_t)PTS * 128 * 2;     // tnet max result
    f16* BIG1 = (f16*)(ws + o); o += (size_t)PTS * 1024 * 2;    // t3 then x4
    f16* X1   = (f16*)(ws + o); o += (size_t)PTS * 64 * 2;
    f16* X2   = (f16*)(ws + o); o += (size_t)PTS * 64 * 2;
    f16* X3   = (f16*)(ws + o); o += (size_t)PTS * 128 * 2;
    f16* CAT  = (f16*)(ws + o); o += (size_t)PTS * 256 * 2;
    int* IDX  = (int*)(ws + o); o += (size_t)PTS * KNN * 4;
    float* XT2 = (float*)(ws + o); o += (size_t)PTS * 3 * 4;
    float* SQ  = (float*)(ws + o); o += (size_t)PTS * 4;
    float* SM  = (float*)(ws + o); o += 65536;
    float* GMAX = SM;            // 4096
    float* FC1  = SM + 4096;     // 2048
    float* FC2  = SM + 6144;     // 1024
    float* T9   = SM + 7168;     // 36
    float* CV   = SM + 7232;     // 256
    float* XG   = SM + 7488;     // 4096
    float* GRAMF = (float*)HBUF;                            // 64 MB alias (per batch)
    f16* H6 = HBUF;                                         // [P,512] f16 (16 MB)
    f16* H7 = (f16*)((char*)HBUF + ((size_t)40 << 20));     // [P,256] f16
    f16* HCAT = EBUF;                                       // [P,2112] f16 (66 MB)

    // packed weight sub-buffers (offsets in halves)
    f16* tw1p = WP + 0;        // 64x32
    f16* tw2p = WP + 2048;     // 128x64
    f16* tw3p = WP + 10240;    // 1024x128
    f16* w1p  = WP + 141312;   // 64x32
    f16* w2p  = WP + 143360;   // 64x128
    f16* w3p  = WP + 151552;   // 128x128
    f16* w5p  = WP + 167936;   // 1024x256
    f16* w6p  = WP + 430080;   // 512x2112
    f16* w7p  = WP + 1511424;  // 256x512
    f16* w8p  = WP + 1642496;  // 16x256 (Co padded 2->16)

    packW(stream, tw1, 64, 6,    tw1p, 64, 32);
    packW(stream, tw2, 128, 64,  tw2p, 128, 64);
    packW(stream, tw3, 1024, 128, tw3p, 1024, 128);
    packW(stream, w1, 64, 6,     w1p, 64, 32);
    packW(stream, w2, 64, 128,   w2p, 64, 128);
    packW(stream, w3, 128, 128,  w3p, 128, 128);
    packW(stream, w5, 1024, 256, w5p, 1024, 256);
    packW(stream, w6, 512, 2112, w6p, 512, 2112);
    packW(stream, w7, 256, 512,  w7p, 256, 512);
    packW(stream, w8, 2, 256,    w8p, 16, 256);

    // ---- Transform_Net ----
    knn3_kernel<<<PTS / 128, 128, 0, stream>>>(x, IDX);
    edge6_kernel<<<XNK / 256, 256, 0, stream>>>(x, IDX, EBUF);
    gemm(stream, EBUF, 32, tw1p, 32, XNK, 64, 32, ts1, tb1, 0, HBUF, nullptr, 64, NP);
    gemm(stream, HBUF, 64, tw2p, 64, XNK, 128, 64, ts2, tb2, 0, EBUF, nullptr, 128, NP);
    maxk_kernel<<<(PTS * 128) / 256, 256, 0, stream>>>(EBUF, 128, HM, 128, 128);
    gemm(stream, HM, 128, tw3p, 128, PTS, 1024, 128, ts3, tb3, 0, BIG1, nullptr, 1024, NP);
    colmax_kernel<<<(NB * 1024) / 256, 256, 0, stream>>>(BIG1, 1024, GMAX, 1024);
    fc_kernel<<<(NB * 512 + 255) / 256, 256, 0, stream>>>(GMAX, 1024, tl1, tls1, tlb1, FC1, 512, 1, NB * 512);
    fc_kernel<<<(NB * 256 + 255) / 256, 256, 0, stream>>>(FC1, 512, tl2, tls2, tlb2, FC2, 256, 1, NB * 256);
    fc_kernel<<<1, 64, 0, stream>>>(FC2, 256, ttw, nullptr, ttb, T9, 9, 0, NB * 9);
    xform_kernel<<<PTS / 256, 256, 0, stream>>>(x, T9, XT2);

    // ---- EdgeConv 1 (3-D coords, transformed) ----
    knn3_kernel<<<PTS / 128, 128, 0, stream>>>(XT2, IDX);
    edge6_kernel<<<XNK / 256, 256, 0, stream>>>(XT2, IDX, EBUF);
    gemm(stream, EBUF, 32, w1p, 32, XNK, 64, 32, s1, b1, 0, HBUF, nullptr, 64, NP);
    maxk_kernel<<<(PTS * 64) / 256, 256, 0, stream>>>(HBUF, 64, X1, 64, 64);
    sqnorm_kernel<<<PTS / 256, 256, 0, stream>>>(X1, 64, 64, SQ);

    // ---- EdgeConv 2 (WMMA Gram kNN on x1) ----
    for (int b = 0; b < NB; ++b) {
        const f16* fb = X1 + (size_t)b * NP * 64;
        gemm(stream, fb, 64, fb, 64, NP, NP, 64, SQ + b * NP, nullptr, 1, nullptr, GRAMF, NP, NP);
        topk_gram_kernel<<<NP / 128, 128, 0, stream>>>(GRAMF, IDX + (size_t)b * NP * KNN);
    }
    edgeF_kernel<<<(XNK * 64) / 256, 256, 0, stream>>>(X1, 64, 64, IDX, EBUF, 128);
    gemm(stream, EBUF, 128, w2p, 128, XNK, 64, 128, s2, b2, 0, HBUF, nullptr, 64, NP);
    maxk_kernel<<<(PTS * 64) / 256, 256, 0, stream>>>(HBUF, 64, X2, 64, 64);
    sqnorm_kernel<<<PTS / 256, 256, 0, stream>>>(X2, 64, 64, SQ);

    // ---- EdgeConv 3 (WMMA Gram kNN on x2) ----
    for (int b = 0; b < NB; ++b) {
        const f16* fb = X2 + (size_t)b * NP * 64;
        gemm(stream, fb, 64, fb, 64, NP, NP, 64, SQ + b * NP, nullptr, 1, nullptr, GRAMF, NP, NP);
        topk_gram_kernel<<<NP / 128, 128, 0, stream>>>(GRAMF, IDX + (size_t)b * NP * KNN);
    }
    edgeF_kernel<<<(XNK * 64) / 256, 256, 0, stream>>>(X2, 64, 64, IDX, EBUF, 128);
    gemm(stream, EBUF, 128, w3p, 128, XNK, 128, 128, s3, b3, 0, HBUF, nullptr, 128, NP);
    maxk_kernel<<<(PTS * 128) / 256, 256, 0, stream>>>(HBUF, 128, X3, 128, 128);

    // ---- Global aggregation + head ----
    cat_kernel<<<(PTS * 256) / 256, 256, 0, stream>>>(X1, X2, X3, CAT);
    gemm(stream, CAT, 256, w5p, 256, PTS, 1024, 256, s5, b5, 0, BIG1, nullptr, 1024, NP);
    colmax_kernel<<<(NB * 1024) / 256, 256, 0, stream>>>(BIG1, 1024, XG, 1024);
    cv_kernel<<<1, 256, 0, stream>>>(cvec, catw, catlb, cats, catbb, CV);
    hcat_kernel<<<(PTS * 2112) / 256, 256, 0, stream>>>(XG, CV, BIG1, HCAT);
    gemm(stream, HCAT, 2112, w6p, 2112, PTS, 512, 2112, s6, b6, 0, H6, nullptr, 512, NP);
    gemm(stream, H6, 512, w7p, 512, PTS, 256, 512, s7, b7, 0, H7, nullptr, 256, NP);
    gemm(stream, H7, 256, w8p, 256, PTS, 16, 256, nullptr, b8, 2, nullptr, out, 0, NP);
}
